// SparKDensfiyBlock_61435212202174
// MI455X (gfx1250) — matrix operand order repaired and checked by reading, not compiled, not run on hardware
//
#include <hip/hip_runtime.h>
#include <hip/hip_bf16.h>
#include <stdint.h>

// SparK densify block: per-position LN over C + mask-token fill + 3x3 conv (SAME) + bias.
// B=64, C_E=768, C_D=512, H=W=28, K=3.
// bf16 channels-last repack (fused LN+mask), then implicit-GEMM conv via
// v_wmma_f32_16x16x32_bf16 with async global->LDS staging, incremental K addressing.

typedef __attribute__((ext_vector_type(16))) __bf16 v16bf;
typedef __attribute__((ext_vector_type(8)))  __bf16 v8bf;
typedef __attribute__((ext_vector_type(8)))  float  v8f;

#define B_    64
#define CE    768
#define CD    512
#define HW_   28
#define NPOS  (B_ * HW_ * HW_)    // 50176
#define KSTEP 32
#define NSTAGE (CE * 9 / KSTEP)   // 216 (24 stages per (kh,kw) segment)

// ---------------------------------------------------------------------------
// Kernel 1: fused LayerNorm(channel) + mask-token fill -> bf16 channels-last X
// ---------------------------------------------------------------------------
__global__ __launch_bounds__(256) void spark_ln(
    const float* __restrict__ bcff, const unsigned char* __restrict__ act,
    const float* __restrict__ mtok, const float* __restrict__ gamma,
    const float* __restrict__ beta, __bf16* __restrict__ X) {
  const int lane = threadIdx.x & 31;
  const int wv   = threadIdx.x >> 5;
  const int p    = blockIdx.x * 8 + wv;          // grid = NPOS/8 exactly
  const int b    = p / (HW_ * HW_);
  const int hw   = p - b * (HW_ * HW_);

  const float* src = bcff + (size_t)b * CE * (HW_ * HW_) + hw;
  float v[24];
  float s = 0.f, s2 = 0.f;
#pragma unroll
  for (int i = 0; i < 24; ++i) {
    float x = src[(size_t)(lane + 32 * i) * (HW_ * HW_)];
    v[i] = x; s += x; s2 += x * x;
  }
#pragma unroll
  for (int off = 16; off; off >>= 1) {
    s  += __shfl_xor(s,  off, 32);
    s2 += __shfl_xor(s2, off, 32);
  }
  const float mean = s * (1.0f / 768.0f);
  const float var  = s2 * (1.0f / 768.0f) - mean * mean;
  const float rs   = rsqrtf(var + 1e-6f);
  const bool active = act[p] != 0;

  __bf16* dst = X + (size_t)p * CE;
#pragma unroll
  for (int i = 0; i < 24; ++i) {
    int c = lane + 32 * i;
    float nv = (v[i] - mean) * rs * gamma[c] + beta[c];
    float o  = active ? nv : mtok[c];
    dst[c] = (__bf16)o;
  }
}

// ---------------------------------------------------------------------------
// Kernel 2: weight repack conv_w[CD][CE][3][3] -> Wb[(kh*3+kw)][co][ci] bf16,
// plus a 256B zero page used for conv padding.
// ---------------------------------------------------------------------------
__global__ __launch_bounds__(256) void spark_wprep(
    const float* __restrict__ w, __bf16* __restrict__ Wb, __bf16* __restrict__ zp) {
  size_t id = (size_t)blockIdx.x * 256 + threadIdx.x;
  if (id < 128) zp[id] = (__bf16)0.0f;
  if (id >= (size_t)9 * CD * CE) return;
  int ci   = (int)(id % CE);
  int t    = (int)(id / CE);
  int co   = t % CD;
  int sidx = t / CD;              // kh*3 + kw
  int kh   = sidx / 3;
  int kw   = sidx - kh * 3;
  Wb[id] = (__bf16)w[(((size_t)co * CE + ci) * 3 + kh) * 3 + kw];
}

// ---------------------------------------------------------------------------
// Kernel 3: implicit-GEMM 3x3 conv via WMMA bf16 -> f32, async LDS staging.
// Block tile: M=112 positions (4 rows x 28) x N=256 out-channels, K steps of 32.
// 8 waves: wave w owns 32 N columns (two 16x16 B fragments), 14 accumulators.
// K addressing is incremental: +64B per stage, segment rollover every 24 stages.
// ---------------------------------------------------------------------------
__device__ __forceinline__ void async_b128(uint32_t lds_off, uint64_t gaddr) {
  asm volatile("global_load_async_to_lds_b128 %0, %1, off"
               :: "v"(lds_off), "v"(gaddr) : "memory");
}

__global__ __launch_bounds__(256) void spark_gemm(
    const __bf16* __restrict__ X, const __bf16* __restrict__ Wb,
    const __bf16* __restrict__ zeroPage, const float* __restrict__ bias,
    float* __restrict__ out) {
  __shared__ __bf16 sA[2][128 * 32];   // M rows (112 used + 16 pad) x 32 K   (8 KB each)
  __shared__ __bf16 sB[2][256 * 32];   // N rows (co) x 32 K                  (16 KB each)

  const int tid  = threadIdx.x;
  const int lane = tid & 31;
  const int wv   = tid >> 5;       // 0..7
  const int l16  = lane & 15;
  const int half = lane >> 4;

  const int bid = blockIdx.x;      // grid = B_ * 7 * 2
  const int b   = bid / 14;
  const int r   = bid - b * 14;
  const int tm  = r >> 1;          // 0..6 -> oh0 = 4*tm
  const int tn  = r & 1;           // 0..1 -> n0 = 256*tn
  const int oh0 = tm * 4;
  const int n0  = tn * 256;

  const uint32_t sAaddr[2] = { (uint32_t)(uintptr_t)&sA[0][0], (uint32_t)(uintptr_t)&sA[1][0] };
  const uint32_t sBaddr[2] = { (uint32_t)(uintptr_t)&sB[0][0], (uint32_t)(uintptr_t)&sB[1][0] };

  // A: 512 16B-chunks (128 rows x 4), 2 per thread. B: 1024 chunks (256 rows x 4), 4/thread.
  int      aoh[2], aow[2], acol[2];
  uint32_t aoff[2];
  uint32_t boff[4];
  uint64_t aga[2], bga[4];
  uint64_t astep[2];
#pragma unroll
  for (int i = 0; i < 2; ++i) {
    int ac   = tid + i * 256;               // 0..511
    int arow = ac >> 2;                     // LDS row 0..127
    int am   = arow > 111 ? 111 : arow;     // clamp pad rows to a valid address
    aoh[i]  = oh0 + am / HW_;
    aow[i]  = am - (am / HW_) * HW_;
    acol[i] = (ac & 3) * 8;
    aoff[i] = (uint32_t)((arow * 32 + (ac & 3) * 8) * 2);
  }
#pragma unroll
  for (int i = 0; i < 4; ++i) {
    int bc  = tid + i * 256;                // 0..1023
    boff[i] = (uint32_t)(((bc >> 2) * 32 + (bc & 3) * 8) * 2);
    // start address for segment 0: Wb[(0*CD + n0 + co)*CE + (bc&3)*8]
    bga[i]  = (uint64_t)(uintptr_t)(Wb + (size_t)(n0 + (bc >> 2)) * CE + (bc & 3) * 8);
  }

  // Segment (kh,kw) state for the issue side of the pipeline.
  int khc = 0, kwc = 0, jrem = 24;

  auto setup_A = [&](int kh, int kw) {
#pragma unroll
    for (int i = 0; i < 2; ++i) {
      int ih = aoh[i] + kh - 1;
      int iw = aow[i] + kw - 1;
      bool inb = ((unsigned)ih < (unsigned)HW_) & ((unsigned)iw < (unsigned)HW_);
      const __bf16* gp = inb
          ? (X + (size_t)((b * HW_ + ih) * HW_ + iw) * CE + acol[i])
          : zeroPage;
      aga[i]   = (uint64_t)(uintptr_t)gp;
      astep[i] = inb ? 64u : 0u;            // OOB lanes keep re-reading the zero page
    }
  };
  setup_A(0, 0);

  auto issue = [&](int bufi) {
    async_b128(sAaddr[bufi] + aoff[0], aga[0]);
    async_b128(sAaddr[bufi] + aoff[1], aga[1]);
    async_b128(sBaddr[bufi] + boff[0], bga[0]);
    async_b128(sBaddr[bufi] + boff[1], bga[1]);
    async_b128(sBaddr[bufi] + boff[2], bga[2]);
    async_b128(sBaddr[bufi] + boff[3], bga[3]);
    aga[0] += astep[0];
    aga[1] += astep[1];
#pragma unroll
    for (int i = 0; i < 4; ++i) bga[i] += 64;   // +32 ci
    if (--jrem == 0) {                           // segment rollover every 24 stages
      jrem = 24;
      kwc++;
      if (kwc == 3) { kwc = 0; khc++; }
      // after 24 bumps bga advanced exactly CE elems; jump to next (kh,kw) plane
#pragma unroll
      for (int i = 0; i < 4; ++i) bga[i] += (uint64_t)(CD - 1) * CE * 2;
      if (khc < 3) setup_A(khc, kwc);
    }
  };

  v8f acc0[7], acc1[7];
#pragma unroll
  for (int s = 0; s < 7; ++s) { acc0[s] = (v8f)0.0f; acc1[s] = (v8f)0.0f; }

  issue(0);                                      // prologue: stage 0 -> buf 0

  for (int ks = 0; ks < NSTAGE; ++ks) {
    const int buf = ks & 1;
    if (ks + 1 < NSTAGE) {
      issue(buf ^ 1);
      asm volatile("s_wait_asynccnt 0x6" ::: "memory");  // this stage's 6 landed
    } else {
      asm volatile("s_wait_asynccnt 0x0" ::: "memory");
    }
    __syncthreads();                             // all waves' tiles in LDS

    // Two B fragments: lane holds column n = l16 (+0 / +16); K = half*16..+15.
    const __bf16* Bp0 = &sB[buf][(wv * 32 + l16) * 32 + half * 16];
    const __bf16* Bp1 = Bp0 + 16 * 32;
    v8bf bl0 = *(const v8bf*)(Bp0);
    v8bf bh0 = *(const v8bf*)(Bp0 + 8);
    v8bf bl1 = *(const v8bf*)(Bp1);
    v8bf bh1 = *(const v8bf*)(Bp1 + 8);
    v16bf bf0 = __builtin_shufflevector(bl0, bh0,
        0, 1, 2, 3, 4, 5, 6, 7, 8, 9, 10, 11, 12, 13, 14, 15);
    v16bf bf1 = __builtin_shufflevector(bl1, bh1,
        0, 1, 2, 3, 4, 5, 6, 7, 8, 9, 10, 11, 12, 13, 14, 15);

#pragma unroll
    for (int s = 0; s < 7; ++s) {
      // A fragment: lane holds row m = s*16 + l16; K = half*16..+15.
      const __bf16* Ap = &sA[buf][(s * 16 + l16) * 32];
      v8bf a0 = *(const v8bf*)(Ap + half * 16);
      v8bf a1 = *(const v8bf*)(Ap + half * 16 + 8);
      v16bf afrag = __builtin_shufflevector(a0, a1,
          0, 1, 2, 3, 4, 5, 6, 7, 8, 9, 10, 11, 12, 13, 14, 15);
      acc0[s] = __builtin_amdgcn_wmma_f32_16x16x32_bf16(
          false, afrag, false, bf0, (short)0, acc0[s], false, false);
      acc1[s] = __builtin_amdgcn_wmma_f32_16x16x32_bf16(
          false, afrag, false, bf1, (short)0, acc1[s], false, false);
    }
    __syncthreads();                             // buffer free before overwrite
  }

  // Store: lane holds column n; VGPR rr holds row m = half*8 + rr.
#pragma unroll
  for (int p = 0; p < 2; ++p) {
    const int n = n0 + wv * 32 + p * 16 + l16;
    const float bv = bias[n];
    const v8f* accp = p ? acc1 : acc0;
#pragma unroll
    for (int s = 0; s < 7; ++s) {
#pragma unroll
      for (int rr = 0; rr < 8; ++rr) {
        int m  = s * 16 + half * 8 + rr;         // 0..111
        int oh = oh0 + m / HW_;
        int ow = m - (m / HW_) * HW_;
        out[(((size_t)b * CD + n) * HW_ + oh) * HW_ + ow] = accp[s][rr] + bv;
      }
    }
  }
}

// ---------------------------------------------------------------------------
extern "C" void kernel_launch(void* const* d_in, const int* in_sizes, int n_in,
                              void* d_out, int out_size, void* d_ws, size_t ws_size,
                              hipStream_t stream) {
  (void)in_sizes; (void)n_in; (void)out_size; (void)ws_size;
  const float*         bcff  = (const float*)d_in[0];
  const unsigned char* act   = (const unsigned char*)d_in[1];
  const float*         mtok  = (const float*)d_in[2];
  const float*         gamma = (const float*)d_in[3];
  const float*         beta  = (const float*)d_in[4];
  const float*         convw = (const float*)d_in[5];
  const float*         convb = (const float*)d_in[6];
  float*               out   = (float*)d_out;

  char* ws = (char*)d_ws;
  __bf16* zp = (__bf16*)ws;                                   // 256 B zero page
  __bf16* X  = (__bf16*)(ws + 256);                           // NPOS*CE bf16
  __bf16* Wb = (__bf16*)(ws + 256 + (size_t)NPOS * CE * 2);   // 9*CD*CE bf16

  spark_ln<<<NPOS / 8, 256, 0, stream>>>(bcff, act, mtok, gamma, beta, X);
  spark_wprep<<<(9 * CD * CE + 255) / 256, 256, 0, stream>>>(convw, Wb, zp);
  spark_gemm<<<B_ * 7 * 2, 256, 0, stream>>>(X, Wb, zp, convb, out);
}
